// KernelNonLocalConv_66314295050716
// MI455X (gfx1250) — compile-verified, matrix operand order checked by reading, not run on hardware
//
#include <hip/hip_runtime.h>
#include <cmath>

typedef __attribute__((ext_vector_type(16))) _Float16 v16h;
typedef __attribute__((ext_vector_type(8)))  _Float16 v8h;
typedef __attribute__((ext_vector_type(8)))  float    v8f;
typedef __attribute__((ext_vector_type(4)))  float    v4f;

#define B_  2
#define C_  64
#define H_  256
#define W_  256
#define CH_ 32
#define E_  8
#define HW_ (H_*W_)

// Combine two 8-half runs (the lane's K-runs k0..k0+7 and k0+16..k0+23) into a
// 16x16x32 f16 WMMA fragment.
__device__ __forceinline__ v16h frag16(const _Float16* lo, const _Float16* hi) {
  v8h l = *(const v8h*)lo;
  v8h h = *(const v8h*)hi;
  return __builtin_shufflevector(l, h, 0,1,2,3,4,5,6,7,8,9,10,11,12,13,14,15);
}

// ---------------------------------------------------------------------------
// Generic conv (1x1 or 3x3, stride 1/2), implicit-GEMM via v_wmma_f32_16x16x32_f16.
// Optional: 2x nearest-neighbor upsample fused into input read, residual add,
// per-batch weight matrices (used for the non-local S @ th GEMM).
// Block = 64 threads (2 waves). Block computes 32 couts x 32 pixels:
//   wave w -> couts [og*32 + w*16, +16); each wave keeps 2 accumulators for the
//   two 16-pixel halves, reusing one A fragment per 2 WMMAs.
// LDS tiles are stored K(cin)-contiguous so every fragment is 2 x ds_load_b128.
// ---------------------------------------------------------------------------
__global__ __launch_bounds__(64)
void conv_wmma(const float* __restrict__ in, const float* __restrict__ wgt,
               const float* __restrict__ bias, const float* __restrict__ resid,
               float* __restrict__ out,
               int Cin, int Cout, int Hsrc, int Wsrc, int Ho, int Wo,
               int ksize, int stride, int upsample, int w_per_batch)
{
  __shared__ _Float16 ilds[3 * 65 * 64];   // [(ky*span + col)*Cin_pad + c]
  __shared__ _Float16 wlds[32 * 576];      // [(m*taps + t)*Cin_pad + c]

  const int taps    = ksize * ksize;
  const int pad     = ksize >> 1;
  const int Cin_pad = (Cin + 31) & ~31;
  const int span    = 31 * stride + ksize;
  const int tiles_x = Wo >> 5;
  const int tile    = blockIdx.x;
  const int ty      = tile / tiles_x;
  const int x0      = (tile % tiles_x) << 5;
  const int og      = blockIdx.y;
  const int bb      = blockIdx.z;
  const int tid     = threadIdx.x;
  const int Hi      = upsample ? (Hsrc << 1) : Hsrc;
  const int Wi      = upsample ? (Wsrc << 1) : Wsrc;

  // ---- stage weights for 32 couts (zero-pad cin / cout), K-contiguous ----
  const long wbase = w_per_batch ? (long)bb * Cout * Cin * taps : 0;
  for (int ii = tid; ii < 32 * taps * Cin_pad; ii += 64) {
    int c  = ii % Cin_pad;
    int mt = ii / Cin_pad;           // m*taps + t
    int m  = mt / taps;
    int t  = mt - m * taps;
    int co = og * 32 + m;
    float v = 0.f;
    if (c < Cin && co < Cout)
      v = wgt[wbase + ((long)co * Cin + c) * taps + t];
    wlds[ii] = (_Float16)v;
  }
  // ---- stage input strip, channel-innermost (conflict-free LDS writes) ----
  for (int ii = tid; ii < ksize * span * Cin_pad; ii += 64) {
    int c  = ii % Cin_pad;
    int rc = ii / Cin_pad;           // r*span + col
    int r  = rc / span;
    int col = rc - r * span;
    int iy = ty * stride - pad + r;
    int ix = x0 * stride - pad + col;
    float v = 0.f;
    if (c < Cin && iy >= 0 && iy < Hi && ix >= 0 && ix < Wi) {
      int sy = upsample ? (iy >> 1) : iy;
      int sx = upsample ? (ix >> 1) : ix;
      v = in[(((long)bb * Cin + c) * Hsrc + sy) * Wsrc + sx];
    }
    ilds[ii] = (_Float16)v;
  }
  __syncthreads();

  const int wave = tid >> 5;
  const int lane = tid & 31;
  const int m15  = lane & 15;
  const int half = lane >> 4;
  const int hoff = half << 3;        // lane's K-run start within a 32-chunk

  v8f acc0 = {0.f,0.f,0.f,0.f,0.f,0.f,0.f,0.f};
  v8f acc1 = {0.f,0.f,0.f,0.f,0.f,0.f,0.f,0.f};

  for (int t = 0; t < taps; ++t) {
    int ky = t / ksize, kx = t - ky * ksize;
    int colb0 = m15 * stride + kx;
    int colb1 = colb0 + (stride << 4);
    int wrow  = ((wave * 16 + m15) * taps + t) * Cin_pad;
    int irow0 = (ky * span + colb0) * Cin_pad;
    int irow1 = (ky * span + colb1) * Cin_pad;
    for (int kc = 0; kc < Cin_pad; kc += 32) {
      int ko = kc + hoff;
      v16h a  = frag16(&wlds[wrow  + ko], &wlds[wrow  + ko + 16]);
      v16h b0 = frag16(&ilds[irow0 + ko], &ilds[irow0 + ko + 16]);
      v16h b1 = frag16(&ilds[irow1 + ko], &ilds[irow1 + ko + 16]);
      acc0 = __builtin_amdgcn_wmma_f32_16x16x32_f16(false, a, false, b0,
                                                    (short)0, acc0, false, false);
      acc1 = __builtin_amdgcn_wmma_f32_16x16x32_f16(false, a, false, b1,
                                                    (short)0, acc1, false, false);
    }
  }

  // ---- store: C/D layout VGPR r -> M = r + 8*half, N = lane&15 ----
  const int xA = x0 + m15;
  const int xB = xA + 16;
  #pragma unroll
  for (int r = 0; r < 8; ++r) {
    int co = og * 32 + wave * 16 + r + hoff;
    if (co < Cout) {
      float bi = bias ? bias[co] : 0.f;
      long rowb = (((long)bb * Cout + co) * Ho + ty) * Wo;
      long iA = rowb + xA, iB = rowb + xB;
      float vA = acc0[r] + bi, vB = acc1[r] + bi;
      if (resid) { vA += resid[iA]; vB += resid[iB]; }
      out[iA] = vA;
      out[iB] = vB;
    }
  }
}

// ---------------------------------------------------------------------------
// S[b, c, c'] = (1/N) * sum_m g[b,c,m] * ph[b,c',m]    (M=N'=64, K=4096)
// One wave per 16x16 tile of S; operands loaded as 16B global_load_b128 runs.
// ---------------------------------------------------------------------------
__global__ __launch_bounds__(32)
void sgemm_wmma(const float* __restrict__ g, const float* __restrict__ ph,
                float* __restrict__ S, int Cc, int Npix, float scale)
{
  const int lane = threadIdx.x;
  const int m15  = lane & 15;
  const int half = lane >> 4;
  const int bb   = blockIdx.z;
  const float* gb = g  + (long)bb * Cc * Npix + (long)(blockIdx.x * 16 + m15) * Npix;
  const float* pb = ph + (long)bb * Cc * Npix + (long)(blockIdx.y * 16 + m15) * Npix;

  v8f acc = {0.f,0.f,0.f,0.f,0.f,0.f,0.f,0.f};
  for (int k0 = 0; k0 < Npix; k0 += 32) {
    int ko = k0 + (half << 3);
    v4f a0 = *(const v4f*)(gb + ko);
    v4f a1 = *(const v4f*)(gb + ko + 4);
    v4f a2 = *(const v4f*)(gb + ko + 16);
    v4f a3 = *(const v4f*)(gb + ko + 20);
    v4f b0 = *(const v4f*)(pb + ko);
    v4f b1 = *(const v4f*)(pb + ko + 4);
    v4f b2 = *(const v4f*)(pb + ko + 16);
    v4f b3 = *(const v4f*)(pb + ko + 20);
    v16h a, b;
    #pragma unroll
    for (int e = 0; e < 4; ++e) {
      a[e]      = (_Float16)a0[e];  a[e + 4]  = (_Float16)a1[e];
      a[e + 8]  = (_Float16)a2[e];  a[e + 12] = (_Float16)a3[e];
      b[e]      = (_Float16)b0[e];  b[e + 4]  = (_Float16)b1[e];
      b[e + 8]  = (_Float16)b2[e];  b[e + 12] = (_Float16)b3[e];
    }
    acc = __builtin_amdgcn_wmma_f32_16x16x32_f16(false, a, false, b,
                                                 (short)0, acc, false, false);
  }
  #pragma unroll
  for (int r = 0; r < 8; ++r) {
    int m = blockIdx.x * 16 + r + (half << 3);
    int n = blockIdx.y * 16 + m15;
    S[((long)bb * Cc + m) * Cc + n] = acc[r] * scale;
  }
}

// ---------------------------------------------------------------------------
// feat[b,c,y,x] = chan[b,c,y,x] * sum_t x[b,c,y+dy,x+dx] * k9[b,t,y,x]
// ---------------------------------------------------------------------------
__global__ __launch_bounds__(256)
void dynfeat_kernel(const float* __restrict__ x, const float* __restrict__ chan,
                    const float* __restrict__ k9, float* __restrict__ feat)
{
  long idx = (long)blockIdx.x * 256 + threadIdx.x;
  if (idx >= (long)B_ * C_ * HW_) return;
  int xw = idx & (W_ - 1);
  long t1 = idx >> 8;
  int yy = t1 & (H_ - 1);
  long t2 = t1 >> 8;
  int c  = t2 & (C_ - 1);
  int bb = (int)(t2 >> 6);

  const float* xc = x + ((long)bb * C_ + c) * HW_;
  const long kb   = (long)bb * 9 * HW_ + (long)yy * W_ + xw;
  float s = 0.f;
  #pragma unroll
  for (int t = 0; t < 9; ++t) {
    int dy = t / 3 - 1, dx = t % 3 - 1;
    int iy = yy + dy, ix = xw + dx;
    if (iy >= 0 && iy < H_ && ix >= 0 && ix < W_)
      s += xc[(long)iy * W_ + ix] * k9[kb + (long)t * HW_];
  }
  feat[idx] = chan[idx] * s;
}

// ---------------------------------------------------------------------------
// Per-pixel 8-expert mixture:
//   rw[e] = sigmoid(w_fc[e,:] . h1 + b_fc[e]);  out[o] = sum_e rw[e]*(W_e h1)[o]
// ---------------------------------------------------------------------------
__global__ __launch_bounds__(256)
void experts_kernel(const float* __restrict__ h1, const float* __restrict__ w_fc,
                    const float* __restrict__ b_fc, const float* __restrict__ w_ex,
                    float* __restrict__ outm)
{
  __shared__ float sfc[E_ * CH_];
  __shared__ float sbf[E_];
  __shared__ float sex[E_ * CH_ * CH_];
  const int tid = threadIdx.x;
  for (int i = tid; i < E_ * CH_; i += 256) sfc[i] = w_fc[i];
  if (tid < E_) sbf[tid] = b_fc[tid];
  for (int i = tid; i < E_ * CH_ * CH_; i += 256) sex[i] = w_ex[i];
  __syncthreads();

  long p = (long)blockIdx.x * 256 + tid;
  if (p >= (long)B_ * HW_) return;
  int bb  = (int)(p >> 16);
  int rem = (int)(p & (HW_ - 1));

  float h[CH_];
  #pragma unroll
  for (int i = 0; i < CH_; ++i)
    h[i] = h1[((long)bb * CH_ + i) * HW_ + rem];

  float acc[CH_];
  #pragma unroll
  for (int o = 0; o < CH_; ++o) acc[o] = 0.f;

  for (int e = 0; e < E_; ++e) {
    float z = sbf[e];
    #pragma unroll
    for (int i = 0; i < CH_; ++i) z += sfc[e * CH_ + i] * h[i];
    float rw = 1.f / (1.f + expf(-z));
    for (int o = 0; o < CH_; ++o) {
      float d = 0.f;
      #pragma unroll
      for (int i = 0; i < CH_; ++i) d += sex[(e * CH_ + o) * CH_ + i] * h[i];
      acc[o] += rw * d;
    }
  }
  #pragma unroll
  for (int o = 0; o < CH_; ++o)
    outm[((long)bb * CH_ + o) * HW_ + rem] = acc[o];
}

// ---------------------------------------------------------------------------
static inline void launch_conv(hipStream_t s, const float* in, const float* w,
                               const float* b, const float* resid, float* out,
                               int Cin, int Cout, int Hsrc, int Wsrc, int Ho, int Wo,
                               int ks, int stride, int ups, int wpb)
{
  dim3 grid((unsigned)((Wo >> 5) * Ho), (unsigned)((Cout + 31) >> 5), (unsigned)B_);
  conv_wmma<<<grid, 64, 0, s>>>(in, w, b, resid, out,
                                Cin, Cout, Hsrc, Wsrc, Ho, Wo, ks, stride, ups, wpb);
}

extern "C" void kernel_launch(void* const* d_in, const int* in_sizes, int n_in,
                              void* d_out, int out_size, void* d_ws, size_t ws_size,
                              hipStream_t stream) {
  // setup_inputs() dict order: x, then params in insertion order
  const float* x      = (const float*)d_in[0];
  const float* w_span = (const float*)d_in[1],  *b_span = (const float*)d_in[2];
  const float* w_ch   = (const float*)d_in[3],  *b_ch   = (const float*)d_in[4];
  const float* w_down = (const float*)d_in[5],  *b_down = (const float*)d_in[6];
  const float* w_d1   = (const float*)d_in[7],  *b_d1   = (const float*)d_in[8];
  const float* w_d2   = (const float*)d_in[9],  *b_d2   = (const float*)d_in[10];
  const float* w_g    = (const float*)d_in[11], *b_g    = (const float*)d_in[12];
  const float* w_th   = (const float*)d_in[13], *b_th   = (const float*)d_in[14];
  const float* w_ph   = (const float*)d_in[15], *b_ph   = (const float*)d_in[16];
  const float* w_wz   = (const float*)d_in[17], *b_wz   = (const float*)d_in[18];
  const float* w_u1   = (const float*)d_in[19], *b_u1   = (const float*)d_in[20];
  const float* w_u2   = (const float*)d_in[21], *b_u2   = (const float*)d_in[22];
  const float* w_up   = (const float*)d_in[23], *b_up   = (const float*)d_in[24];
  const float* w_c1   = (const float*)d_in[25], *b_c1   = (const float*)d_in[26];
  const float* w_c2   = (const float*)d_in[27], *b_c2   = (const float*)d_in[28];
  const float* w_fc   = (const float*)d_in[29], *b_fc   = (const float*)d_in[30];
  const float* w_ex   = (const float*)d_in[31];
  float* out = (float*)d_out;

  // ---- workspace carving (floats), with lifetime-safe aliasing ----
  float* ws = (float*)d_ws;
  size_t off = 0;
  auto alloc = [&](size_t n) { float* p = ws + off; off += n; return p; };
  float* spanb = alloc((size_t)B_ * 9  * HW_);        // span features
  float* chanb = alloc((size_t)B_ * C_ * HW_);        // chan (lives until dynfeat)
  float* bigb  = alloc((size_t)B_ * C_ * HW_);        // ktmp (full-res) then u2
  float* k1u1  = alloc((size_t)B_ * C_ * 128 * 128);  // k_d1 then u1
  float* k2b   = alloc((size_t)B_ * C_ * 64 * 64);    // k at H/4 (residual for wz)
  float* gbuf  = alloc((size_t)B_ * C_ * 4096);
  float* thbuf = alloc((size_t)B_ * C_ * 4096);
  float* phbuf = alloc((size_t)B_ * C_ * 4096);
  float* Sbuf  = alloc((size_t)B_ * C_ * C_);
  float* ybuf  = alloc((size_t)B_ * C_ * 4096);
  float* k4b   = alloc((size_t)B_ * C_ * 4096);
  float* k9b   = alloc((size_t)B_ * 9  * HW_);
  float* featb = alloc((size_t)B_ * C_ * HW_);
  float* h1b   = alloc((size_t)B_ * CH_ * HW_);
  float* mixb  = alloc((size_t)B_ * CH_ * HW_);
  (void)ws_size; (void)in_sizes; (void)n_in; (void)out_size;

  // ---- dynamic-kernel generation path ----
  launch_conv(stream, x, w_span, b_span, nullptr, spanb, 64,  9, 256, 256, 256, 256, 3, 1, 0, 0);
  launch_conv(stream, x, w_ch,   b_ch,   nullptr, chanb, 64, 64, 256, 256, 256, 256, 3, 1, 0, 0);
  launch_conv(stream, spanb, w_down, b_down, nullptr, bigb,  9, 64, 256, 256, 256, 256, 1, 1, 0, 0);
  launch_conv(stream, bigb,  w_d1,   b_d1,   nullptr, k1u1, 64, 64, 256, 256, 128, 128, 3, 2, 0, 0);
  launch_conv(stream, k1u1,  w_d2,   b_d2,   nullptr, k2b,  64, 64, 128, 128,  64,  64, 3, 2, 0, 0);

  // ---- non-local block, algebraically fused: y = ((g phT)/N) th ----
  launch_conv(stream, k2b, w_g,  b_g,  nullptr, gbuf,  64, 64, 64, 64, 64, 64, 1, 1, 0, 0);
  launch_conv(stream, k2b, w_th, b_th, nullptr, thbuf, 64, 64, 64, 64, 64, 64, 1, 1, 0, 0);
  launch_conv(stream, k2b, w_ph, b_ph, nullptr, phbuf, 64, 64, 64, 64, 64, 64, 1, 1, 0, 0);
  sgemm_wmma<<<dim3(4, 4, B_), 32, 0, stream>>>(gbuf, phbuf, Sbuf, 64, 4096, 1.0f / 4096.0f);
  launch_conv(stream, thbuf, Sbuf, nullptr, nullptr, ybuf, 64, 64, 64, 64, 64, 64, 1, 1, 0, 1);
  launch_conv(stream, ybuf,  w_wz, b_wz, k2b,     k4b,  64, 64, 64, 64, 64, 64, 1, 1, 0, 0);

  // ---- upsample back to full resolution, form 9-tap per-pixel kernels ----
  launch_conv(stream, k4b,  w_u1, b_u1, nullptr, k1u1, 64, 64,  64,  64, 128, 128, 3, 1, 1, 0);
  launch_conv(stream, k1u1, w_u2, b_u2, nullptr, bigb, 64, 64, 128, 128, 256, 256, 3, 1, 1, 0);
  launch_conv(stream, bigb, w_up, b_up, nullptr, k9b,  64,  9, 256, 256, 256, 256, 1, 1, 0, 0);

  // ---- fused unfold * dynamic kernel ----
  dynfeat_kernel<<<(unsigned)(((long)B_ * C_ * HW_ + 255) / 256), 256, 0, stream>>>(
      x, chanb, k9b, featb);

  // ---- mixed-expert branch ----
  launch_conv(stream, featb, w_c1, b_c1, nullptr, h1b, 64, 32, 256, 256, 256, 256, 3, 1, 0, 0);
  experts_kernel<<<(unsigned)(((long)B_ * HW_ + 255) / 256), 256, 0, stream>>>(
      h1b, w_fc, b_fc, w_ex, mixb);
  launch_conv(stream, mixb, w_c2, b_c2, featb, out, 32, 64, 256, 256, 256, 256, 3, 1, 0, 0);
}
